// Model_13469017440656
// MI455X (gfx1250) — matrix operand order repaired and checked by reading, not compiled, not run on hardware
//
#include <hip/hip_runtime.h>

// ---------------- CDNA5 WMMA types ----------------
typedef __attribute__((ext_vector_type(16))) __bf16 v16bf;
typedef __attribute__((ext_vector_type(8)))  __bf16 v8bf;
typedef __attribute__((ext_vector_type(8)))  float  v8f;

#define KPAD_MAX 1024   // max K = round32(9*112) = 1024 -> 32KB LDS im2col tile

// =================================================================
// Generic implicit-GEMM conv (3x3 or 1x1 "linear") via WMMA bf16.
// One wave computes (16 output sites) x (NT*16 output channels):
// the A fragment is reused NT times and the NT accumulator chains
// are independent, so the XDL pipe can issue WMMAs back-to-back
// without hitting the WMMA->WMMA RAW hazard.
//   in   : fp32 NHWC, channel-padded to CinPad (pad region holds zeros)
//   wpk  : bf16 weights pre-packed in exact B-fragment order
//   out[site*ostride + ooff + n] = (acc [+bias]) * omask[site]
// =================================================================
template<int NT>
__global__ __launch_bounds__(32)
void conv_wmma(const float* __restrict__ in, const __bf16* __restrict__ wpk,
               const float* __restrict__ omask, const float* __restrict__ bias,
               float* __restrict__ out,
               int B, int H, int W, int OH, int OW,
               int CinPad, int taps, int Kpad, int stride, int pad,
               int CoutReal, int ostride, int ooff)
{
    __shared__ __bf16 lA[16 * KPAD_MAX];

    const int  lane   = threadIdx.x;
    const long nsites = (long)B * OH * OW;
    const long site0  = (long)blockIdx.x * 16;
    const int  KReal  = taps * CinPad;

    // ---- stage im2col patch (16 sites x Kpad) into LDS as bf16 ----
    for (int idx = lane; idx < 16 * Kpad; idx += 32) {
        int  mrow = idx / Kpad;
        int  kk   = idx - mrow * Kpad;
        float v = 0.0f;
        long site = site0 + mrow;
        if (site < nsites && kk < KReal) {
            int b  = (int)(site / ((long)OH * OW));
            int r  = (int)(site % ((long)OH * OW));
            int oy = r / OW, ox = r % OW;
            int tap = kk / CinPad;
            int ci  = kk - tap * CinPad;
            int ky  = tap / 3, kx = tap % 3;
            int iy  = oy * stride + ky - pad;
            int ix  = ox * stride + kx - pad;
            if (iy >= 0 && iy < H && ix >= 0 && ix < W)
                v = in[(((long)b * H + iy) * W + ix) * CinPad + ci];
        }
        lA[mrow * Kpad + kk] = (__bf16)v;
    }
    __syncthreads();

    // ---- K loop: v_wmma_f32_16x16x32_bf16, NT independent accumulators ----
    v8f acc[NT] = {};
    const int  nk = Kpad >> 5;
    const long wbase = (long)blockIdx.y * NT * nk * 512;
    const int  mrow = lane & 15;
    const int  hi   = lane >> 4;

    for (int kc = 0; kc < nk; ++kc) {
        const int kb = kc << 5;
        // A fragment: lane<16 -> K {0..15}; lane>=16 -> K {16..31}
        v8bf alo = *(const v8bf*)&lA[mrow * Kpad + kb + hi * 16];
        v8bf ahi = *(const v8bf*)&lA[mrow * Kpad + kb + hi * 16 + 8];
        v16bf a  = __builtin_shufflevector(alo, ahi,
                     0,1,2,3,4,5,6,7,8,9,10,11,12,13,14,15);
        #pragma unroll
        for (int nt = 0; nt < NT; ++nt) {
            // B fragment: one contiguous 32-byte load (pre-packed)
            v16bf bmat = *(const v16bf*)(wpk + wbase +
                             ((long)nt * nk * 32 + (long)kc * 32 + lane) * 16);
            acc[nt] = __builtin_amdgcn_wmma_f32_16x16x32_bf16(
                          false, a, false, bmat, (short)0, acc[nt], false, false);
        }
    }

    // ---- store: C/D layout -> VGPR r holds M = r + hi*8, N = lane&15 ----
    #pragma unroll
    for (int nt = 0; nt < NT; ++nt) {
        const int n = ((blockIdx.y * NT + nt) << 4) + mrow;
        #pragma unroll
        for (int r = 0; r < 8; ++r) {
            long site = site0 + r + hi * 8;
            if (site < nsites && n < CoutReal) {
                float v = acc[nt][r];
                if (bias)  v += bias[n];
                if (omask) v *= omask[site];
                out[site * (long)ostride + ooff + n] = v;
            }
        }
    }
}

// =================================================================
// Pack fp32 OIHW (or [N,K] for taps==1) weights into bf16 B-fragment
// order: pk[((nt*nk + kc)*32 + lane)*16 + j], where
//   lane<16:  N = nt*16+lane,    K = kc*32 + j
//   lane>=16: N = nt*16+lane-16, K = kc*32 + 16 + j
// =================================================================
__global__ void pack_w(const float* __restrict__ w, __bf16* __restrict__ pk,
                       int Cin, int Cout, int taps, int CinPad, int Kpad, int CoutPad)
{
    const int  nk    = Kpad / 32;
    const long total = (long)(CoutPad / 16) * nk * 512;
    for (long i = blockIdx.x * (long)blockDim.x + threadIdx.x; i < total;
         i += (long)gridDim.x * blockDim.x) {
        int  j    = (int)(i & 15);
        int  lane = (int)((i >> 4) & 31);
        long rest = i >> 9;
        int  kc   = (int)(rest % nk);
        int  nt   = (int)(rest / nk);
        int  n    = nt * 16 + (lane & 15);
        int  k    = kc * 32 + (lane >> 4) * 16 + j;
        float v = 0.0f;
        if (n < Cout && k < taps * CinPad) {
            int tap = k / CinPad, ci = k % CinPad;
            if (ci < Cin) v = w[((long)n * Cin + ci) * taps + tap];
        }
        pk[i] = (__bf16)v;
    }
}

// ---------------- element-wise / pooling / BN helpers ----------------
__global__ void zero_f32(float* __restrict__ p, long n)
{
    for (long i = blockIdx.x * (long)blockDim.x + threadIdx.x; i < n;
         i += (long)gridDim.x * blockDim.x) p[i] = 0.0f;
}

// NCHW (B,3,H,W) * mask -> NHWC padded to 16 channels
__global__ void to_nhwc(const float* __restrict__ x, const float* __restrict__ m,
                        float* __restrict__ out, int B, int HW)
{
    long total = (long)B * HW * 16;
    for (long i = blockIdx.x * (long)blockDim.x + threadIdx.x; i < total;
         i += (long)gridDim.x * blockDim.x) {
        int  c = (int)(i & 15);
        long s = i >> 4;
        long b = s / HW, r = s % HW;
        out[i] = (c < 3) ? x[(b * 3 + c) * HW + r] * m[s] : 0.0f;
    }
}

// 3x3/2 VALID max pool on (B,HW) mask
__global__ void pool_mask(const float* __restrict__ in, float* __restrict__ out,
                          int B, int H, int W, int OH, int OW)
{
    long total = (long)B * OH * OW;
    for (long i = blockIdx.x * (long)blockDim.x + threadIdx.x; i < total;
         i += (long)gridDim.x * blockDim.x) {
        int b = (int)(i / ((long)OH * OW));
        int r = (int)(i % ((long)OH * OW));
        int oy = r / OW, ox = r % OW;
        const float* p = in + (long)b * H * W;
        float v = 0.0f;
        for (int ky = 0; ky < 3; ++ky)
            for (int kx = 0; kx < 3; ++kx)
                v = fmaxf(v, p[(oy * 2 + ky) * W + ox * 2 + kx]);
        out[i] = v;
    }
}

// 3x3/2 VALID max pool on NHWC activations, re-masked at coarse scale
__global__ void pool_act(const float* __restrict__ in, const float* __restrict__ mc,
                         float* __restrict__ out, int B, int H, int W,
                         int OH, int OW, int Cpad)
{
    long total = (long)B * OH * OW * Cpad;
    for (long i = blockIdx.x * (long)blockDim.x + threadIdx.x; i < total;
         i += (long)gridDim.x * blockDim.x) {
        int  c = (int)(i % Cpad);
        long s = i / Cpad;
        int b = (int)(s / ((long)OH * OW));
        int r = (int)(s % ((long)OH * OW));
        int oy = r / OW, ox = r % OW;
        float v = 0.0f;
        for (int ky = 0; ky < 3; ++ky)
            for (int kx = 0; kx < 3; ++kx)
                v = fmaxf(v, in[(((long)b * H + oy * 2 + ky) * W + ox * 2 + kx) * Cpad + c]);
        out[i] = v * mc[s];
    }
}

// masked per-channel sums: stats[c]=Σxm, stats[Cpad+c]=Σx²m, stats[2*Cpad]=Σm
__global__ void bn_stats(const float* __restrict__ x, const float* __restrict__ m,
                         float* __restrict__ st, long nsites, int C, int Cpad)
{
    int  c   = blockIdx.y;
    long gid = blockIdx.x * (long)blockDim.x + threadIdx.x;
    long gsz = (long)gridDim.x * blockDim.x;
    float s = 0.0f, s2 = 0.0f, cm = 0.0f;
    for (long i = gid; i < nsites; i += gsz) {
        float mv = m[i];
        float xv = x[i * Cpad + c];
        s  += xv * mv;
        s2 += xv * xv * mv;
        cm += mv;
    }
    atomicAdd(&st[c], s);
    atomicAdd(&st[Cpad + c], s2);
    if (c == 0) atomicAdd(&st[2 * Cpad], cm);
}

// y = relu((x-mu)*rsqrt(var+eps)*g + b) * m ; pad channels forced to 0
__global__ void bn_apply(float* __restrict__ x, const float* __restrict__ m,
                         const float* __restrict__ st, const float* __restrict__ g,
                         const float* __restrict__ bb, long nsites, int C, int Cpad)
{
    long total = nsites * Cpad;
    float cnt  = st[2 * Cpad];
    for (long i = blockIdx.x * (long)blockDim.x + threadIdx.x; i < total;
         i += (long)gridDim.x * blockDim.x) {
        int  c = (int)(i % Cpad);
        long s = i / Cpad;
        if (c >= C) { x[i] = 0.0f; continue; }
        float mu  = st[c] / cnt;
        float var = st[Cpad + c] / cnt - mu * mu;
        float y   = (x[i] - mu) * rsqrtf(var + 1e-4f) * g[c] + bb[c];
        x[i] = fmaxf(y, 0.0f) * m[s];
    }
}

// transposed conv 3x3 / stride 2 / VALID: coarse CxC -> fine FxF, re-masked fine
__global__ void deconv_kernel(const float* __restrict__ z, const float* __restrict__ w,
                              const float* __restrict__ mf, float* __restrict__ out,
                              int B, int F, int C, int bch, int aoff, int ostride)
{
    long total = (long)B * F * F * bch;
    for (long i = blockIdx.x * (long)blockDim.x + threadIdx.x; i < total;
         i += (long)gridDim.x * blockDim.x) {
        int  co = (int)(i % bch);
        long s  = i / bch;
        int b = (int)(s / ((long)F * F));
        int r = (int)(s % ((long)F * F));
        int y = r / F, x2 = r % F;
        float acc = 0.0f;
        for (int ky = 0; ky < 3; ++ky) {
            int yy = y - ky;
            if (yy < 0 || (yy & 1)) continue;
            int iy = yy >> 1; if (iy >= C) continue;
            for (int kx = 0; kx < 3; ++kx) {
                int xx = x2 - kx;
                if (xx < 0 || (xx & 1)) continue;
                int ix = xx >> 1; if (ix >= C) continue;
                const float* zz = z + (((long)b * C + iy) * C + ix) * 16;
                const float* ww = w + (long)co * bch * 9 + ky * 3 + kx;
                for (int ci = 0; ci < bch; ++ci) acc += zz[ci] * ww[(long)ci * 9];
            }
        }
        out[s * (long)ostride + aoff + co] = acc * mf[s];
    }
}

// ---------------- host-side helpers ----------------
static inline long cdivl(long a, long b) { return (a + b - 1) / b; }
static inline int  pad16(int x) { return (x + 15) & ~15; }
static inline int  rnd32(int x) { return (x + 31) & ~31; }
static inline int  pickNT(int ntiles) {
    if (ntiles % 4 == 0) return 4;
    if (ntiles % 2 == 0) return 2;
    return 1;
}

static void launch_conv(int ntiles, long nsites, hipStream_t stream,
                        const float* in, const __bf16* wpk,
                        const float* omask, const float* bias, float* out,
                        int B, int H, int W, int OH, int OW,
                        int CinPad, int taps, int Kpad, int stride, int pad,
                        int CoutReal, int ostride, int ooff)
{
    int NT = pickNT(ntiles);
    dim3 g((unsigned)cdivl(nsites, 16), ntiles / NT);
    switch (NT) {
    case 4:
        conv_wmma<4><<<g, 32, 0, stream>>>(in, wpk, omask, bias, out,
            B, H, W, OH, OW, CinPad, taps, Kpad, stride, pad, CoutReal, ostride, ooff);
        break;
    case 2:
        conv_wmma<2><<<g, 32, 0, stream>>>(in, wpk, omask, bias, out,
            B, H, W, OH, OW, CinPad, taps, Kpad, stride, pad, CoutReal, ostride, ooff);
        break;
    default:
        conv_wmma<1><<<g, 32, 0, stream>>>(in, wpk, omask, bias, out,
            B, H, W, OH, OW, CinPad, taps, Kpad, stride, pad, CoutReal, ostride, ooff);
        break;
    }
}

extern "C" void kernel_launch(void* const* d_in, const int* in_sizes, int n_in,
                              void* d_out, int out_size, void* d_ws, size_t ws_size,
                              hipStream_t stream)
{
    (void)in_sizes; (void)n_in; (void)out_size; (void)ws_size;
    const float* x     = (const float*)d_in[0];
    const float* mask0 = (const float*)d_in[1];
    const int B = 512;
    const int dims[6] = {63, 31, 15, 7, 3, 1};

    // deterministic bump allocator into d_ws
    char*  ws  = (char*)d_ws;
    size_t off = 0;
    auto alloc = [&](size_t bytes) -> void* {
        void* p = ws + off;
        off += (bytes + 255) & ~(size_t)255;
        return p;
    };
    auto grid1 = [&](long n, int bs) { return dim3((unsigned)cdivl(n, bs)); };

    // ---- mask pyramid 63 -> 31 -> 15 -> 7 -> 3 -> 1 ----
    const float* mL[6];
    mL[0] = mask0;
    for (int s = 1; s < 6; ++s) {
        float* mm = (float*)alloc((size_t)B * dims[s] * dims[s] * 4);
        long n = (long)B * dims[s] * dims[s];
        pool_mask<<<grid1(n, 256), 256, 0, stream>>>(mL[s - 1], mm, B,
            dims[s - 1], dims[s - 1], dims[s], dims[s]);
        mL[s] = mm;
    }

    // ---- activation / scratch buffers ----
    size_t actBytes = (size_t)B * 63 * 63 * 32 * sizeof(float);
    float* actA = (float*)alloc(actBytes);
    float* actB = (float*)alloc(actBytes);
    size_t zBytes = (size_t)B * 31 * 31 * 16 * sizeof(float);
    float* zA = (float*)alloc(zBytes);
    float* zB = (float*)alloc(zBytes);
    float* hfin  = (float*)alloc((size_t)B * 128 * sizeof(float));
    float* stats = (float*)alloc(1024 * sizeof(float));

    // ---- input NCHW*mask -> NHWC pad16 ----
    to_nhwc<<<grid1((long)B * 63 * 63 * 16, 256), 256, 0, stream>>>(
        x, mask0, actA, B, 63 * 63);

    static const int BCFG[10][3] = {
        {3,16,8},{24,16,8},{24,32,8},{40,32,8},{40,48,16},
        {64,48,16},{64,64,16},{80,64,16},{80,96,16},{112,96,16}};

    float* cur = actA; float* nxt = actB;
    int scale = 0;
    for (int blk = 0; blk < 10; ++blk) {
        if (blk == 2 || blk == 4 || blk == 6 || blk == 8) {   // 'MP'
            scale++;
            int Cpad = pad16(BCFG[blk][0]);
            int H = dims[scale - 1], O = dims[scale];
            pool_act<<<grid1((long)B * O * O * Cpad, 256), 256, 0, stream>>>(
                cur, mL[scale], nxt, B, H, H, O, O, Cpad);
            float* t = cur; cur = nxt; nxt = t;
        }
        const int cin = BCFG[blk][0], a = BCFG[blk][1], bch = BCFG[blk][2];
        const int CinPad = pad16(cin), CpadOut = pad16(a + bch);
        const int F = dims[scale], Cc = dims[scale + 1];
        const float* w1 = (const float*)d_in[2 + blk * 8 + 0];
        const float* w2 = (const float*)d_in[2 + blk * 8 + 1];
        const float* g1 = (const float*)d_in[2 + blk * 8 + 2];
        const float* b1 = (const float*)d_in[2 + blk * 8 + 3];
        const float* w3 = (const float*)d_in[2 + blk * 8 + 4];
        const float* w4 = (const float*)d_in[2 + blk * 8 + 5];
        const float* g2 = (const float*)d_in[2 + blk * 8 + 6];
        const float* b2 = (const float*)d_in[2 + blk * 8 + 7];

        const long nFine   = (long)B * F * F;
        const long nCoarse = (long)B * Cc * Cc;

        // --- branch 1: submanifold conv cin->a, s1 SAME, *mfine -> nxt[:,0:a)
        {
            int Kpad = rnd32(9 * CinPad);
            __bf16* pk = (__bf16*)alloc((size_t)(a / 16) * (Kpad / 32) * 512 * 2);
            pack_w<<<grid1((long)(a / 16) * (Kpad / 32) * 512, 256), 256, 0, stream>>>(
                w1, pk, cin, a, 9, CinPad, Kpad, a);
            launch_conv(a / 16, nFine, stream, cur, pk, mL[scale], nullptr, nxt,
                B, F, F, F, F, CinPad, 9, Kpad, 1, 1, a, CpadOut, 0);
        }
        // --- branch 2a: conv cin->b, s2 VALID, *mcoarse -> zA
        {
            int Kpad = rnd32(9 * CinPad);
            __bf16* pk = (__bf16*)alloc((size_t)(Kpad / 32) * 512 * 2);
            pack_w<<<grid1((long)(Kpad / 32) * 512, 256), 256, 0, stream>>>(
                w2, pk, cin, bch, 9, CinPad, Kpad, 16);
            launch_conv(1, nCoarse, stream, cur, pk, mL[scale + 1], nullptr, zA,
                B, F, F, Cc, Cc, CinPad, 9, Kpad, 2, 0, 16, 16, 0);
        }
        // --- branch 2b: BN+ReLU on zA over active coarse sites
        {
            zero_f32<<<1, 256, 0, stream>>>(stats, 2 * 16 + 1);
            bn_stats<<<dim3(64, bch), 256, 0, stream>>>(zA, mL[scale + 1], stats,
                nCoarse, bch, 16);
            bn_apply<<<grid1(nCoarse * 16, 256), 256, 0, stream>>>(
                zA, mL[scale + 1], stats, g1, b1, nCoarse, bch, 16);
        }
        // --- branch 2c: submanifold conv b->b at coarse scale -> zB
        {
            int Kpad = rnd32(9 * 16);   // 160
            __bf16* pk = (__bf16*)alloc((size_t)(Kpad / 32) * 512 * 2);
            pack_w<<<grid1((long)(Kpad / 32) * 512, 256), 256, 0, stream>>>(
                w3, pk, bch, bch, 9, 16, Kpad, 16);
            launch_conv(1, nCoarse, stream, zA, pk, mL[scale + 1], nullptr, zB,
                B, Cc, Cc, Cc, Cc, 16, 9, Kpad, 1, 1, 16, 16, 0);
        }
        // --- branch 2d: deconv 3x3/2 back to fine sites -> nxt[:, a:a+b)
        deconv_kernel<<<grid1(nFine * bch, 256), 256, 0, stream>>>(
            zB, w4, mL[scale], nxt, B, F, Cc, bch, a, CpadOut);
        // --- join: BN+ReLU over concat(a+b), masked fine, pads zeroed
        {
            zero_f32<<<1, 256, 0, stream>>>(stats, 2 * CpadOut + 1);
            bn_stats<<<dim3(128, a + bch), 256, 0, stream>>>(nxt, mL[scale], stats,
                nFine, a + bch, CpadOut);
            bn_apply<<<grid1(nFine * CpadOut, 256), 256, 0, stream>>>(
                nxt, mL[scale], stats, g2, b2, nFine, a + bch, CpadOut);
        }
        float* t = cur; cur = nxt; nxt = t;
    }

    // ---- final conv 112->128, 3x3 s2 VALID (3x3 -> 1x1), *m1 ----
    const float* wf  = (const float*)d_in[82];
    const float* gf  = (const float*)d_in[83];
    const float* bfp = (const float*)d_in[84];
    const float* lw  = (const float*)d_in[85];
    const float* lb  = (const float*)d_in[86];
    {
        int Kpad = rnd32(9 * 112);   // 1024
        __bf16* pk = (__bf16*)alloc((size_t)8 * (Kpad / 32) * 512 * 2);
        pack_w<<<grid1((long)8 * (Kpad / 32) * 512, 256), 256, 0, stream>>>(
            wf, pk, 112, 128, 9, 112, Kpad, 128);
        launch_conv(8, (long)B, stream, cur, pk, mL[5], nullptr, hfin,
            B, 3, 3, 1, 1, 112, 9, Kpad, 2, 0, 128, 128, 0);
    }
    {
        zero_f32<<<1, 256, 0, stream>>>(stats, 2 * 128 + 1);
        bn_stats<<<dim3(8, 128), 256, 0, stream>>>(hfin, mL[5], stats, B, 128, 128);
        bn_apply<<<grid1((long)B * 128, 256), 256, 0, stream>>>(
            hfin, mL[5], stats, gf, bfp, B, 128, 128);
    }
    // ---- classifier: [512,128] @ [128,3755] + bias -> d_out ----
    // pad N to 3776 (236 tiles, divisible by 4) so the biggest GEMM runs NT=4
    {
        int Kpad = 128, CoPad = 3776;
        __bf16* pk = (__bf16*)alloc((size_t)(CoPad / 16) * (Kpad / 32) * 512 * 2);
        pack_w<<<grid1((long)(CoPad / 16) * (Kpad / 32) * 512, 256), 256, 0, stream>>>(
            lw, pk, 128, 3755, 1, 128, Kpad, CoPad);
        launch_conv(CoPad / 16, (long)B, stream, hfin, pk, nullptr, lb, (float*)d_out,
            B, 1, 1, 1, 1, 128, 1, Kpad, 1, 0, 3755, 3755, 0);
    }
}
